// Attention_22900765622520
// MI455X (gfx1250) — matrix operand order repaired and checked by reading, not compiled, not run on hardware
//
#include <hip/hip_runtime.h>
#include <stdint.h>

// ---------------------------------------------------------------------------
// Problem constants (from reference): B=2, S=2048, HID=2048, NH=16, NKV=4,
// HD=128, GROUPS=4.  All GEMM/attention math on bf16 WMMA, fp32 accumulate.
// ---------------------------------------------------------------------------
#define BATCH   2
#define SEQ     2048
#define HID     2048
#define NH      16
#define NKV     4
#define HD      128
#define NQKV    3072            // NH*HD + 2*NKV*HD
#define SCALE_F 0.08838834764831845f   // 1/sqrt(128)

typedef __attribute__((ext_vector_type(16))) __bf16 v16bf;
typedef __attribute__((ext_vector_type(8)))  float  v8f;

// fp32 -> bf16 (round-to-nearest-even)
__device__ inline unsigned short f2bf(float f) {
    unsigned int u = __float_as_uint(f);
    unsigned int r = u + 0x7FFFu + ((u >> 16) & 1u);
    return (unsigned short)(r >> 16);
}

__device__ inline v8f vzero() {
    v8f z;
#pragma unroll
    for (int i = 0; i < 8; i++) z[i] = 0.0f;
    return z;
}

// ---------------------------------------------------------------------------
// CDNA5 async global->LDS DMA (ASYNCcnt-tracked).  Low 32 bits of a generic
// LDS pointer are the LDS byte offset (ISA 10.2).
// ---------------------------------------------------------------------------
__device__ inline void async_ld_b128(void* lds, const void* gaddr) {
    unsigned loff = (unsigned)(size_t)lds;
    asm volatile("global_load_async_to_lds_b128 %0, %1, off"
                 :: "v"(loff), "v"(gaddr) : "memory");
}

__device__ inline void wait_async0() {
    asm volatile("s_wait_asynccnt 0" ::: "memory");
}

// A-operand loader: halves [0..7] at p, halves [8..15] at p+16 elements
// (16-bit A matrix 16x32 layout: lanes<16 get K 0-7 / 16-23, lanes>=16 K 8-15 / 24-31;
//  caller bakes the +8 for the upper half-wave into p)
__device__ inline v16bf op_a(const unsigned short* p) {
    union { v16bf v; uint4 q[2]; } u;
    u.q[0] = *reinterpret_cast<const uint4*>(p);
    u.q[1] = *reinterpret_cast<const uint4*>(p + 16);
    return u.v;
}

// B-operand loader: this lane's N column, K rows 0-7 / 16-23 at p (caller bakes +8 rows for upper half-wave)
__device__ inline v16bf op_b(const unsigned short* p, int stride) {
    union { v16bf v; unsigned short e[16]; } u;
#pragma unroll
    for (int s = 0; s < 8; s++) {
        u.e[s]     = p[s * stride];
        u.e[s + 8] = p[(16 + s) * stride];
    }
    return u.v;
}

__device__ inline v8f wmma_bf16(v16bf a, v16bf b, v8f c) {
    return __builtin_amdgcn_wmma_f32_16x16x32_bf16(
        /*neg_a=*/false, a, /*neg_b=*/false, b,
        /*c_mod=*/(short)0, c, /*reuse_a=*/false, /*reuse_b=*/false);
}

// ---------------------------------------------------------------------------
// fp32 -> bf16 bulk convert
// ---------------------------------------------------------------------------
__global__ void cvt_kernel(const float* __restrict__ in,
                           unsigned short* __restrict__ out, size_t n) {
    size_t i = (size_t)blockIdx.x * blockDim.x + threadIdx.x;
    if (i < n) out[i] = f2bf(in[i]);
}

// Pack a weight [K x ncols] into columns [off, off+ncols) of Wcat [K x NQKV]
__global__ void wcat_kernel(const float* __restrict__ in,
                            unsigned short* __restrict__ wcat,
                            int ncols, int off, size_t n) {
    size_t i = (size_t)blockIdx.x * blockDim.x + threadIdx.x;
    if (i >= n) return;
    int c = (int)(i % ncols);
    size_t k = i / ncols;
    wcat[k * NQKV + off + c] = f2bf(in[i]);
}

// ---------------------------------------------------------------------------
// Generic bf16 GEMM: C[M,N] = A[M,K] * Bm[K,N], fp32 out.
// Workgroup = 8 waves, tile 128(M) x 64(N), K-step 64.
// Double-buffered LDS staging via async global->LDS DMA: tile (s+1) streams
// in while tile s feeds the WMMAs; one barrier per K-step.  Operand loads
// are batched so ds_load latency overlaps the WMMA chain.
// ---------------------------------------------------------------------------
#define ASTR 80   // LDS row stride (bf16 elems) for A tile, 160B (16B aligned)
#define BSTR 80

__global__ __launch_bounds__(256)
void gemm_bf16_kernel(const unsigned short* __restrict__ A,
                      const unsigned short* __restrict__ Bm,
                      float* __restrict__ C, int M, int N, int K) {
    __shared__ unsigned short As[2][128 * ASTR];
    __shared__ unsigned short Bs[2][64 * BSTR];

    const int tid  = threadIdx.x;
    const int lane = tid & 31;
    const int wave = tid >> 5;
    const int hh   = lane >> 4;
    const int l15  = lane & 15;
    const int n0   = blockIdx.x * 64;
    const int m0   = blockIdx.y * 128;

    v8f acc[4];
#pragma unroll
    for (int i = 0; i < 4; i++) acc[i] = vzero();

    // issue async DMA for one K-step tile pair into buffer `buf`
    auto stage = [&](int buf, int k0) {
        // A tile: 128 rows x 64 bf16 = 1024 x 16B chunks, 4 per thread
#pragma unroll
        for (int i = 0; i < 4; i++) {
            int ch = tid + i * 256;
            int r = ch >> 3, c = ch & 7;
            async_ld_b128(&As[buf][r * ASTR + c * 8],
                          A + (size_t)(m0 + r) * K + k0 + c * 8);
        }
        // B tile: 64 K-rows x 64 bf16 = 512 chunks, 2 per thread
#pragma unroll
        for (int i = 0; i < 2; i++) {
            int ch = tid + i * 256;
            int r = ch >> 3, c = ch & 7;
            async_ld_b128(&Bs[buf][r * BSTR + c * 8],
                          Bm + (size_t)(k0 + r) * N + n0 + c * 8);
        }
    };

    const int nstage = K / 64;
    stage(0, 0);

    for (int s = 0; s < nstage; s++) {
        wait_async0();        // this wave's DMA into buf (s&1) done
        __syncthreads();      // everyone's DMA done; prev compute done
        if (s + 1 < nstage) stage((s + 1) & 1, (s + 1) * 64);

        const unsigned short* Asb = As[s & 1];
        const unsigned short* Bsb = Bs[s & 1];

        // batch operand loads, then WMMA chains (hides ds_load latency)
        v16bf a0 = op_a(Asb + (wave * 16 + l15) * ASTR + 0  + hh * 8);
        v16bf a1 = op_a(Asb + (wave * 16 + l15) * ASTR + 32 + hh * 8);
        v16bf bq[4];
#pragma unroll
        for (int c4 = 0; c4 < 4; c4++)
            bq[c4] = op_b(Bsb + (0 + hh * 8) * BSTR + c4 * 16 + l15, BSTR);
#pragma unroll
        for (int c4 = 0; c4 < 4; c4++)
            acc[c4] = wmma_bf16(a0, bq[c4], acc[c4]);
#pragma unroll
        for (int c4 = 0; c4 < 4; c4++)
            bq[c4] = op_b(Bsb + (32 + hh * 8) * BSTR + c4 * 16 + l15, BSTR);
#pragma unroll
        for (int c4 = 0; c4 < 4; c4++)
            acc[c4] = wmma_bf16(a1, bq[c4], acc[c4]);
    }

    // C/D layout: lane half selects M 0-7 / 8-15, vgpr idx = row-in-half, l15 = N
#pragma unroll
    for (int c4 = 0; c4 < 4; c4++) {
#pragma unroll
        for (int v = 0; v < 8; v++) {
            int m = m0 + wave * 16 + hh * 8 + v;
            int n = n0 + c4 * 16 + l15;
            C[(size_t)m * N + n] = acc[c4][v];
        }
    }
}

// ---------------------------------------------------------------------------
// RoPE on Q and K (fp32 in, bf16 out).  Q written [b,h][s][d] row-major
// (A operand form); K written TRANSPOSED [b,kvh][d][s] (B operand form).
// One thread per (b, s, head-slot 0..19, j 0..63).
// ---------------------------------------------------------------------------
__global__ void rope_qk_kernel(const float* __restrict__ QKV,
                               unsigned short* __restrict__ Qo,
                               unsigned short* __restrict__ Kt) {
    size_t idx = (size_t)blockIdx.x * blockDim.x + threadIdx.x;
    int j = (int)(idx & 63);
    size_t t = idx >> 6;
    int hh = (int)(t % 20);
    t /= 20;
    int s = (int)(t % SEQ);
    int b = (int)(t / SEQ);

    float inv = __powf(10000.0f, -(float)j * (1.0f / 64.0f));
    float sn, cs;
    __sincosf((float)s * inv, &sn, &cs);

    const float* row = QKV + ((size_t)(b * SEQ + s)) * NQKV;
    if (hh < NH) {
        float x1 = row[hh * HD + j];
        float x2 = row[hh * HD + j + 64];
        unsigned short* dst = Qo + ((size_t)(b * NH + hh) * SEQ + s) * HD;
        dst[j]      = f2bf(x1 * cs - x2 * sn);
        dst[j + 64] = f2bf(x2 * cs + x1 * sn);
    } else {
        int kh = hh - NH;
        float x1 = row[NH * HD + kh * HD + j];
        float x2 = row[NH * HD + kh * HD + j + 64];
        size_t base = (size_t)(b * NKV + kh) * HD;
        Kt[(base + j)      * SEQ + s] = f2bf(x1 * cs - x2 * sn);
        Kt[(base + j + 64) * SEQ + s] = f2bf(x2 * cs + x1 * sn);
    }
}

// V: plain fp32->bf16, row-major [b,kvh][s][d] (already B-operand form for P*V)
__global__ void vconv_kernel(const float* __restrict__ QKV,
                             unsigned short* __restrict__ Vb) {
    size_t idx = (size_t)blockIdx.x * blockDim.x + threadIdx.x;
    int d = (int)(idx & (HD - 1));
    int s = (int)((idx >> 7) & (SEQ - 1));
    int kh = (int)((idx >> 18) & 3);
    int b = (int)(idx >> 20);
    float x = QKV[((size_t)(b * SEQ + s)) * NQKV + (NH + NKV) * HD + kh * HD + d];
    Vb[((size_t)(b * NKV + kh) * SEQ + s) * HD + d] = f2bf(x);
}

// ---------------------------------------------------------------------------
// Flash attention.  Grid (S/128, NH, B); 8 waves, each owns 16 q rows.
// Per 32-key block: S = Q*K^T (8 wmma), online softmax (shuffle reductions),
// P relayout via private LDS slice, O += P*V (8 wmma).
// K/V tiles are double-buffered and streamed by async global->LDS DMA so the
// next block's tiles load while this block computes.  Operand loads batched.
// ---------------------------------------------------------------------------
#define KTSTR 40    // Kt tile row stride (80B rows, 16B aligned, low conflict)
#define VSTR  136   // V tile row stride (272B rows)
#define PSTR  40

__global__ __launch_bounds__(256)
void flash_kernel(const unsigned short* __restrict__ Q,
                  const unsigned short* __restrict__ Kt,
                  const unsigned short* __restrict__ Vb,
                  unsigned short* __restrict__ O) {
    __shared__ unsigned short Kts[2][HD * KTSTR];    // [d=128][32 keys]
    __shared__ unsigned short Vs[2][32 * VSTR];      // [32 keys][128 d]
    __shared__ unsigned short Ps[8 * 16 * PSTR];     // per-wave 16x32 P tiles

    const int tid  = threadIdx.x;
    const int lane = tid & 31;
    const int wave = tid >> 5;
    const int hh   = lane >> 4;
    const int l15  = lane & 15;
    const int qb   = blockIdx.x;
    const int h    = blockIdx.y;
    const int b    = blockIdx.z;
    const int kvh  = h >> 2;    // GROUPS = 4

    const unsigned short* Qbase = Q  + ((size_t)(b * NH  + h)   * SEQ) * HD;
    const unsigned short* Kbase = Kt + ((size_t)(b * NKV + kvh) * HD)  * SEQ;
    const unsigned short* Vbase = Vb + ((size_t)(b * NKV + kvh) * SEQ) * HD;

    // issue async DMA for key-block kb into buffer `buf`
    auto stageKV = [&](int buf, int kb) {
        // Kt tile: 128 d-rows x 32 keys -> 512 x 16B chunks, 2 per thread
#pragma unroll
        for (int i = 0; i < 2; i++) {
            int ch = tid + i * 256;
            int r = ch >> 2, c = ch & 3;
            async_ld_b128(&Kts[buf][r * KTSTR + c * 8],
                          Kbase + (size_t)r * SEQ + kb * 32 + c * 8);
        }
        // V tile: 32 key-rows x 128 d -> 512 chunks, 2 per thread
#pragma unroll
        for (int i = 0; i < 2; i++) {
            int ch = tid + i * 256;
            int r = ch >> 4, c = ch & 15;
            async_ld_b128(&Vs[buf][r * VSTR + c * 8],
                          Vbase + (size_t)(kb * 32 + r) * HD + c * 8);
        }
    };

    stageKV(0, 0);

    // Preload this wave's 16x128 Q tile as 4 A operands
    const int qrow = qb * 128 + wave * 16 + l15;
    v16bf qa[4];
#pragma unroll
    for (int kc = 0; kc < 4; kc++)
        qa[kc] = op_a(Qbase + (size_t)qrow * HD + kc * 32 + hh * 8);

    v8f o[8];
#pragma unroll
    for (int i = 0; i < 8; i++) o[i] = vzero();
    float mrow[8], lrow[8];
#pragma unroll
    for (int v = 0; v < 8; v++) { mrow[v] = -3.0e38f; lrow[v] = 0.0f; }

    unsigned short* Pw = Ps + wave * 16 * PSTR;

    for (int kb = 0; kb < SEQ / 32; kb++) {
        wait_async0();        // DMA into buf (kb&1) complete (this wave)
        __syncthreads();      // complete for all waves; prev block done
        if (kb + 1 < SEQ / 32) stageKV((kb + 1) & 1, kb + 1);

        const unsigned short* Ktb = Kts[kb & 1];
        const unsigned short* Vtb = Vs[kb & 1];

        // ---- scores: two 16x16 tiles over K-dim 128; batch 8 operand loads
        v16bf bk[8];
#pragma unroll
        for (int kc = 0; kc < 4; kc++) {
            bk[kc * 2]     = op_b(Ktb + (kc * 32 + hh * 8) * KTSTR + 0 + l15, KTSTR);
            bk[kc * 2 + 1] = op_b(Ktb + (kc * 32 + hh * 8) * KTSTR + 16 + l15, KTSTR);
        }
        v8f s0 = vzero(), s1 = vzero();
#pragma unroll
        for (int kc = 0; kc < 4; kc++) {
            s0 = wmma_bf16(qa[kc], bk[kc * 2],     s0);
            s1 = wmma_bf16(qa[kc], bk[kc * 2 + 1], s1);
        }

        // ---- online softmax ----
        float rmax[8];
#pragma unroll
        for (int v = 0; v < 8; v++) {
            s0[v] *= SCALE_F; s1[v] *= SCALE_F;
            rmax[v] = fmaxf(s0[v], s1[v]);
        }
#pragma unroll
        for (int mk = 1; mk < 16; mk <<= 1)
#pragma unroll
            for (int v = 0; v < 8; v++)
                rmax[v] = fmaxf(rmax[v], __shfl_xor(rmax[v], mk, 32));

        float corr[8];
#pragma unroll
        for (int v = 0; v < 8; v++) {
            float mn = fmaxf(mrow[v], rmax[v]);
            corr[v] = __expf(mrow[v] - mn);
            mrow[v] = mn;
        }
#pragma unroll
        for (int nc = 0; nc < 8; nc++)
#pragma unroll
            for (int v = 0; v < 8; v++)
                o[nc][v] *= corr[v];

        float rsum[8];
#pragma unroll
        for (int v = 0; v < 8; v++) {
            float p0 = __expf(s0[v] - mrow[v]);
            float p1 = __expf(s1[v] - mrow[v]);
            rsum[v] = p0 + p1;
            // C-layout -> LDS: row = hh*8+v, key col = l15 / 16+l15
            Pw[(hh * 8 + v) * PSTR + l15]      = f2bf(p0);
            Pw[(hh * 8 + v) * PSTR + 16 + l15] = f2bf(p1);
        }
#pragma unroll
        for (int mk = 1; mk < 16; mk <<= 1)
#pragma unroll
            for (int v = 0; v < 8; v++)
                rsum[v] += __shfl_xor(rsum[v], mk, 32);
#pragma unroll
        for (int v = 0; v < 8; v++)
            lrow[v] = lrow[v] * corr[v] + rsum[v];

        // Re-read P as A operand (same wave; compiler inserts ds waits)
        v16bf pa = op_a(Pw + l15 * PSTR + hh * 8);

        // ---- O += P * V : batch 8 V operand loads, then WMMA chain ----
        v16bf vbq[8];
#pragma unroll
        for (int nc = 0; nc < 8; nc++)
            vbq[nc] = op_b(Vtb + (hh * 8) * VSTR + nc * 16 + l15, VSTR);
#pragma unroll
        for (int nc = 0; nc < 8; nc++)
            o[nc] = wmma_bf16(pa, vbq[nc], o[nc]);
    }

    // ---- epilogue: normalize and store bf16 O [b*S+s][NH*HD] ----
#pragma unroll
    for (int v = 0; v < 8; v++) {
        float inv = 1.0f / lrow[v];
#pragma unroll
        for (int nc = 0; nc < 8; nc++) o[nc][v] *= inv;
    }
#pragma unroll
    for (int nc = 0; nc < 8; nc++) {
#pragma unroll
        for (int v = 0; v < 8; v++) {
            int q = qb * 128 + wave * 16 + hh * 8 + v;
            int col = h * HD + nc * 16 + l15;
            O[(size_t)(b * SEQ + q) * (NH * HD) + col] = f2bf(o[nc][v]);
        }
    }
}

// ---------------------------------------------------------------------------
// Host-side launch
// ---------------------------------------------------------------------------
extern "C" void kernel_launch(void* const* d_in, const int* in_sizes, int n_in,
                              void* d_out, int out_size, void* d_ws, size_t ws_size,
                              hipStream_t stream) {
    const float* X  = (const float*)d_in[0];
    const float* Wq = (const float*)d_in[1];
    const float* Wk = (const float*)d_in[2];
    const float* Wv = (const float*)d_in[3];
    const float* Wo = (const float*)d_in[4];
    float* out = (float*)d_out;

    char* ws = (char*)d_ws;
    const size_t SZ_XBF  = (size_t)BATCH * SEQ * HID * 2;        // 16 MB
    const size_t SZ_WCAT = (size_t)HID * NQKV * 2;               // 12 MB
    const size_t SZ_WOBF = (size_t)HID * HID * 2;                // 8 MB
    const size_t SZ_QKV  = (size_t)BATCH * SEQ * NQKV * 4;       // 48 MB
    const size_t SZ_QBF  = (size_t)BATCH * NH * SEQ * HD * 2;    // 16 MB
    const size_t SZ_KT   = (size_t)BATCH * NKV * SEQ * HD * 2;   // 4 MB
    const size_t SZ_VB   = SZ_KT;                                // 4 MB

    unsigned short* Xbf  = (unsigned short*)(ws);
    unsigned short* Wcat = (unsigned short*)(ws + SZ_XBF);
    unsigned short* Wobf = (unsigned short*)(ws + SZ_XBF + SZ_WCAT);
    float*          QKV  = (float*)(ws + SZ_XBF + SZ_WCAT + SZ_WOBF);
    unsigned short* Qbf  = (unsigned short*)(ws + SZ_XBF + SZ_WCAT + SZ_WOBF + SZ_QKV);
    unsigned short* Ktb  = (unsigned short*)((char*)Qbf + SZ_QBF);
    unsigned short* Vbb  = (unsigned short*)((char*)Ktb + SZ_KT);
    unsigned short* Obf  = (unsigned short*)((char*)Vbb + SZ_VB);

    const size_t nX = (size_t)BATCH * SEQ * HID;                 // 8388608
    cvt_kernel<<<(unsigned)((nX + 255) / 256), 256, 0, stream>>>(X, Xbf, nX);
    const size_t nWo = (size_t)HID * HID;
    cvt_kernel<<<(unsigned)((nWo + 255) / 256), 256, 0, stream>>>(Wo, Wobf, nWo);

    const size_t nWq = (size_t)HID * (NH * HD);
    const size_t nWk = (size_t)HID * (NKV * HD);
    wcat_kernel<<<(unsigned)((nWq + 255) / 256), 256, 0, stream>>>(Wq, Wcat, NH * HD, 0, nWq);
    wcat_kernel<<<(unsigned)((nWk + 255) / 256), 256, 0, stream>>>(Wk, Wcat, NKV * HD, NH * HD, nWk);
    wcat_kernel<<<(unsigned)((nWk + 255) / 256), 256, 0, stream>>>(Wv, Wcat, NKV * HD, (NH + NKV) * HD, nWk);

    // QKV = X * Wcat : [4096 x 2048] * [2048 x 3072]
    gemm_bf16_kernel<<<dim3(NQKV / 64, (BATCH * SEQ) / 128), 256, 0, stream>>>(
        Xbf, Wcat, QKV, BATCH * SEQ, NQKV, HID);

    const size_t nRope = (size_t)BATCH * SEQ * (NH + NKV) * 64;  // 5242880
    rope_qk_kernel<<<(unsigned)(nRope / 256), 256, 0, stream>>>(QKV, Qbf, Ktb);
    const size_t nV = (size_t)BATCH * NKV * SEQ * HD;            // 2097152
    vconv_kernel<<<(unsigned)(nV / 256), 256, 0, stream>>>(QKV, Vbb);

    flash_kernel<<<dim3(SEQ / 128, NH, BATCH), 256, 0, stream>>>(Qbf, Ktb, Vbb, Obf);

    // out = O * Wo : [4096 x 2048] * [2048 x 2048]
    gemm_bf16_kernel<<<dim3(HID / 64, (BATCH * SEQ) / 128), 256, 0, stream>>>(
        Obf, Wobf, out, BATCH * SEQ, HID, HID);
}
